// Transformer_35837207118383
// MI455X (gfx1250) — matrix operand order repaired and checked by reading, not compiled, not run on hardware
//
#include <hip/hip_runtime.h>

typedef unsigned short u16;
typedef unsigned int   u32;
typedef __attribute__((ext_vector_type(16))) _Float16 v16h;
typedef __attribute__((ext_vector_type(8)))  float    v8f;

#define Vv 32000
#define Ss 2048
#define Bb 2
#define Ee 1024
#define Hh 16
#define HDd 64
#define Ff 4096
#define Ll 4

union U16f { v16h h; u16 u[16]; };

__device__ __forceinline__ u16 f2h(float f) {
    union { _Float16 h; u16 u; } c; c.h = (_Float16)f; return c.u;
}

// gfx1250 async global->LDS staging path (ASYNCcnt), guarded so the file
// still compiles (register-staged fallback) if the builtin is absent.
#if defined(__gfx1250__) && __has_builtin(__builtin_amdgcn_global_load_async_to_lds_b128)
#define USE_ASYNC_LDS 1
#else
#define USE_ASYNC_LDS 0
#endif

#if USE_ASYNC_LDS
#define AS1 __attribute__((address_space(1)))
#define AS3 __attribute__((address_space(3)))
typedef __attribute__((ext_vector_type(4))) int i32x4;
__device__ __forceinline__ void wait_asynccnt0() {
#if __has_builtin(__builtin_amdgcn_s_wait_asynccnt)
    __builtin_amdgcn_s_wait_asynccnt(0);
#else
    asm volatile("s_wait_asynccnt 0" ::: "memory");
#endif
}
#endif

// ---------------- convert f32 -> f16 (plain) ----------------
__global__ void k_conv(const float* __restrict__ in, u16* __restrict__ out, int n) {
    int i = blockIdx.x * 256 + threadIdx.x;
    if (i < n) out[i] = f2h(in[i]);
}

// ---------------- convert f32 (K x N) -> f16 transposed (N x K) ----------------
__global__ void k_tconv(const float* __restrict__ in, u16* __restrict__ out, int K, int N) {
    int i = blockIdx.x * 256 + threadIdx.x;
    int total = K * N;
    if (i < total) {
        int k = i / N, n = i - k * N;
        out[(size_t)n * K + k] = f2h(in[i]);
    }
}

// ---------------- embedding: h = word_embed[x] + pos_embed ----------------
__global__ void k_embed(const int* __restrict__ x, const float* __restrict__ we,
                        const float* __restrict__ pos, float* __restrict__ h,
                        u16* __restrict__ hh) {
    int t = blockIdx.x;              // 0..B*S-1
    int s = t & (Ss - 1);
    int tok = x[t];
    int tid = threadIdx.x;
#pragma unroll
    for (int j = 0; j < Ee / 256; ++j) {
        int i = tid + j * 256;
        float v = we[(size_t)tok * Ee + i] + pos[(size_t)s * Ee + i];
        h[(size_t)t * Ee + i] = v;
        hh[(size_t)t * Ee + i] = f2h(v);
    }
}

// ---------------- LayerNorm + residual: h = LN(z)*g+b + h  (writes f32 + f16) -------
__global__ __launch_bounds__(256) void k_ln_res(const float* __restrict__ z,
                                                const float* __restrict__ g,
                                                const float* __restrict__ bta,
                                                float* __restrict__ h,
                                                u16* __restrict__ hh) {
    __shared__ float red[256];
    int row = blockIdx.x, tid = threadIdx.x;
    const float* zr = z + (size_t)row * Ee;
    float v[4]; float s = 0.f;
#pragma unroll
    for (int j = 0; j < 4; ++j) { v[j] = zr[tid + j * 256]; s += v[j]; }
    red[tid] = s; __syncthreads();
    for (int st = 128; st > 0; st >>= 1) { if (tid < st) red[tid] += red[tid + st]; __syncthreads(); }
    float mu = red[0] * (1.0f / Ee); __syncthreads();
    float d2 = 0.f;
#pragma unroll
    for (int j = 0; j < 4; ++j) { float d = v[j] - mu; d2 += d * d; }
    red[tid] = d2; __syncthreads();
    for (int st = 128; st > 0; st >>= 1) { if (tid < st) red[tid] += red[tid + st]; __syncthreads(); }
    float rstd = rsqrtf(red[0] * (1.0f / Ee) + 1e-6f);
#pragma unroll
    for (int j = 0; j < 4; ++j) {
        int i = tid + j * 256;
        float o = (v[j] - mu) * rstd * g[i] + bta[i] + h[(size_t)row * Ee + i];
        h[(size_t)row * Ee + i] = o;
        hh[(size_t)row * Ee + i] = f2h(o);
    }
}

// ---------------- WMMA GEMM: C[M,N] = act(A[M,K] @ Bt[N,K]^T + bias) ----------------
// 128x128 block tile, 4 waves (128 thr), 64x64 wave tile, BK=32,
// double-buffered LDS with async (or register-staged) prefetch.
#define BM 128
#define BN 128
#define BK 32
#define LDSK 40   // BK + 8 pad; row pitch 80B keeps 16B chunks aligned

__global__ __launch_bounds__(128) void k_gemm(const u16* __restrict__ A,
                                              const u16* __restrict__ Bt,
                                              float* __restrict__ Cf,
                                              u16* __restrict__ Ch,
                                              const float* __restrict__ bias,
                                              int M, int N, int K, int relu) {
    __shared__ u16 As[2][BM * LDSK];
    __shared__ u16 Bs[2][BN * LDSK];

    int tid = threadIdx.x;
    int wid = tid >> 5, lane = tid & 31;
    int l15 = lane & 15, hi = lane >> 4;
    int bm = blockIdx.y * BM, bn = blockIdx.x * BN;
    int wm = (wid >> 1) * 64, wn = (wid & 1) * 64;
    int srow = tid;                       // one 32-half row per thread per matrix

    const u16* gA = A  + (size_t)(bm + srow) * K;
    const u16* gB = Bt + (size_t)(bn + srow) * K;

    v8f acc[4][4];
#pragma unroll
    for (int mi = 0; mi < 4; ++mi)
#pragma unroll
        for (int ni = 0; ni < 4; ++ni)
#pragma unroll
            for (int r = 0; r < 8; ++r) acc[mi][ni][r] = 0.f;

    const int nk = K / BK;

#if USE_ASYNC_LDS
    // ---- async prologue: tile 0 -> buf 0
#pragma unroll
    for (int c = 0; c < 4; ++c) {
        __builtin_amdgcn_global_load_async_to_lds_b128(
            (AS1 i32x4*)(void*)(gA + c * 8),
            (AS3 i32x4*)(void*)&As[0][srow * LDSK + c * 8], 0, 0);
        __builtin_amdgcn_global_load_async_to_lds_b128(
            (AS1 i32x4*)(void*)(gB + c * 8),
            (AS3 i32x4*)(void*)&Bs[0][srow * LDSK + c * 8], 0, 0);
    }
    wait_asynccnt0();
    __syncthreads();
#else
    // ---- register-staged prologue: tile 0 -> buf 0
    {
        const uint4* ga = (const uint4*)gA;
        const uint4* gb = (const uint4*)gB;
        uint4 ra[4], rb[4];
#pragma unroll
        for (int c = 0; c < 4; ++c) { ra[c] = ga[c]; rb[c] = gb[c]; }
#pragma unroll
        for (int c = 0; c < 4; ++c) {
            *(uint4*)&As[0][srow * LDSK + c * 8] = ra[c];
            *(uint4*)&Bs[0][srow * LDSK + c * 8] = rb[c];
        }
    }
    __syncthreads();
#endif

    for (int it = 0; it < nk; ++it) {
        int buf = it & 1;
#if USE_ASYNC_LDS
        if (it + 1 < nk) {
            int kt = (it + 1) * BK;
#pragma unroll
            for (int c = 0; c < 4; ++c) {
                __builtin_amdgcn_global_load_async_to_lds_b128(
                    (AS1 i32x4*)(void*)(gA + kt + c * 8),
                    (AS3 i32x4*)(void*)&As[buf ^ 1][srow * LDSK + c * 8], 0, 0);
                __builtin_amdgcn_global_load_async_to_lds_b128(
                    (AS1 i32x4*)(void*)(gB + kt + c * 8),
                    (AS3 i32x4*)(void*)&Bs[buf ^ 1][srow * LDSK + c * 8], 0, 0);
            }
        }
#else
        uint4 ra[4], rb[4];
        if (it + 1 < nk) {
            int kt = (it + 1) * BK;
            const uint4* ga = (const uint4*)(gA + kt);
            const uint4* gb = (const uint4*)(gB + kt);
#pragma unroll
            for (int c = 0; c < 4; ++c) { ra[c] = ga[c]; rb[c] = gb[c]; }
        }
#endif
        // ---- fragments + 16 WMMAs from current buffer
        v16h af[4];
#pragma unroll
        for (int mi = 0; mi < 4; ++mi) {
            const u16* ap = &As[buf][(wm + mi * 16 + l15) * LDSK];
            U16f f;
#pragma unroll
            for (int j = 0; j < 8; ++j) {
                f.u[j]     = ap[hi * 8 + j];
                f.u[8 + j] = ap[16 + hi * 8 + j];
            }
            af[mi] = f.h;
        }
        v16h bfv[4];
#pragma unroll
        for (int ni = 0; ni < 4; ++ni) {
            const u16* bp = &Bs[buf][(wn + ni * 16 + l15) * LDSK];
            U16f f;
#pragma unroll
            for (int j = 0; j < 16; ++j) f.u[j] = bp[(j < 8 ? hi * 8 + j : 16 + hi * 8 + (j - 8))];
            bfv[ni] = f.h;
        }
#pragma unroll
        for (int mi = 0; mi < 4; ++mi)
#pragma unroll
            for (int ni = 0; ni < 4; ++ni)
                acc[mi][ni] = __builtin_amdgcn_wmma_f32_16x16x32_f16(
                    false, af[mi], false, bfv[ni], (short)0, acc[mi][ni], false, false);

#if USE_ASYNC_LDS
        if (it + 1 < nk) wait_asynccnt0();
#else
        if (it + 1 < nk) {
#pragma unroll
            for (int c = 0; c < 4; ++c) {
                *(uint4*)&As[buf ^ 1][srow * LDSK + c * 8] = ra[c];
                *(uint4*)&Bs[buf ^ 1][srow * LDSK + c * 8] = rb[c];
            }
        }
#endif
        __syncthreads();
    }

#pragma unroll
    for (int mi = 0; mi < 4; ++mi)
#pragma unroll
        for (int ni = 0; ni < 4; ++ni) {
            int n = bn + wn + ni * 16 + l15;
            float bz = bias ? bias[n] : 0.f;
#pragma unroll
            for (int r = 0; r < 8; ++r) {
                int m = bm + wm + mi * 16 + r + hi * 8;
                float v = acc[mi][ni][r] + bz;
                if (relu) v = fmaxf(v, 0.f);
                if (Ch) Ch[(size_t)m * N + n] = f2h(v);
                else    Cf[(size_t)m * N + n] = v;
            }
        }
}

// ---------------- causal flash attention (one wave = 32 q rows, one (b,h)) ----------
__global__ __launch_bounds__(32) void k_attn(const u16* __restrict__ qkv,
                                             u16* __restrict__ attn) {
    __shared__ u16 pbuf[32 * 40];
    const int rs = 3 * Ee;                 // 3072 halfs per token row
    int lane = threadIdx.x, l15 = lane & 15, hi = lane >> 4;
    int qb = blockIdx.x * 32;
    int bh = blockIdx.y, bIdx = bh >> 4, hIdx = bh & 15;
    const u16* base = qkv + (size_t)bIdx * Ss * rs;
    int koffh = hIdx * 192, qoffh = koffh + 64, voffh = koffh + 128;

    // preload q A-frags: [mi][d-chunk]
    v16h qf[2][2];
#pragma unroll
    for (int mi = 0; mi < 2; ++mi) {
        const u16* qp = base + (size_t)(qb + mi * 16 + l15) * rs + qoffh;
#pragma unroll
        for (int dc = 0; dc < 2; ++dc) {
            U16f f;
#pragma unroll
            for (int j = 0; j < 8; ++j) {
                f.u[j]     = qp[dc * 32 + hi * 8 + j];
                f.u[8 + j] = qp[dc * 32 + 16 + hi * 8 + j];
            }
            qf[mi][dc] = f.h;
        }
    }

    v8f acc[2][4];
    float mr[2][8], lr[2][8];
#pragma unroll
    for (int mi = 0; mi < 2; ++mi)
#pragma unroll
        for (int r = 0; r < 8; ++r) { mr[mi][r] = -1e30f; lr[mi][r] = 0.f; }
#pragma unroll
    for (int mi = 0; mi < 2; ++mi)
#pragma unroll
        for (int ni = 0; ni < 4; ++ni)
#pragma unroll
            for (int r = 0; r < 8; ++r) acc[mi][ni][r] = 0.f;

    for (int kc = 0; kc <= qb + 31; kc += 32) {
        // ---- scores: 32q x 32k via 8 WMMAs
        v8f sc[2][2];
#pragma unroll
        for (int mi = 0; mi < 2; ++mi)
#pragma unroll
            for (int nj = 0; nj < 2; ++nj)
#pragma unroll
                for (int r = 0; r < 8; ++r) sc[mi][nj][r] = 0.f;
#pragma unroll
        for (int nj = 0; nj < 2; ++nj) {
            const u16* kr = base + (size_t)(kc + nj * 16 + l15) * rs + koffh;
#pragma unroll
            for (int dc = 0; dc < 2; ++dc) {
                U16f f;
#pragma unroll
                for (int j = 0; j < 16; ++j) f.u[j] = kr[dc * 32 + (j < 8 ? hi * 8 + j : 16 + hi * 8 + (j - 8))];
                sc[0][nj] = __builtin_amdgcn_wmma_f32_16x16x32_f16(
                    false, qf[0][dc], false, f.h, (short)0, sc[0][nj], false, false);
                sc[1][nj] = __builtin_amdgcn_wmma_f32_16x16x32_f16(
                    false, qf[1][dc], false, f.h, (short)0, sc[1][nj], false, false);
            }
        }
        // ---- online softmax per m-tile
#pragma unroll
        for (int mi = 0; mi < 2; ++mi) {
            float rm[8];
#pragma unroll
            for (int r = 0; r < 8; ++r) {
                int srow = qb + mi * 16 + r + hi * 8;
#pragma unroll
                for (int nj = 0; nj < 2; ++nj) {
                    int kp = kc + nj * 16 + l15;
                    float v = sc[mi][nj][r] * 0.125f;        // 1/sqrt(64)
                    if (kp > srow) v = -1e30f;
                    sc[mi][nj][r] = v;
                }
                rm[r] = fmaxf(sc[mi][0][r], sc[mi][1][r]);
            }
#pragma unroll
            for (int off = 1; off < 16; off <<= 1)
#pragma unroll
                for (int r = 0; r < 8; ++r)
                    rm[r] = fmaxf(rm[r], __shfl_xor(rm[r], off, 32));
            float corr[8], rsum[8];
#pragma unroll
            for (int r = 0; r < 8; ++r) {
                float mnew = fmaxf(mr[mi][r], rm[r]);
                corr[r] = __expf(mr[mi][r] - mnew);
                mr[mi][r] = mnew;
                float p0 = __expf(sc[mi][0][r] - mnew);
                float p1 = __expf(sc[mi][1][r] - mnew);
                sc[mi][0][r] = p0; sc[mi][1][r] = p1;
                rsum[r] = p0 + p1;
            }
#pragma unroll
            for (int off = 1; off < 16; off <<= 1)
#pragma unroll
                for (int r = 0; r < 8; ++r)
                    rsum[r] += __shfl_xor(rsum[r], off, 32);
#pragma unroll
            for (int r = 0; r < 8; ++r) lr[mi][r] = lr[mi][r] * corr[r] + rsum[r];
#pragma unroll
            for (int ni = 0; ni < 4; ++ni)
#pragma unroll
                for (int r = 0; r < 8; ++r) acc[mi][ni][r] *= corr[r];
            // C-layout probs -> LDS (f16), to be re-read in A-layout
#pragma unroll
            for (int nj = 0; nj < 2; ++nj)
#pragma unroll
                for (int r = 0; r < 8; ++r)
                    pbuf[(mi * 16 + r + hi * 8) * 40 + nj * 16 + l15] = f2h(sc[mi][nj][r]);
        }
        __syncthreads();
        // ---- P @ V
        v16h pf[2];
#pragma unroll
        for (int mi = 0; mi < 2; ++mi) {
            const u16* pp = pbuf + (mi * 16 + l15) * 40;
            U16f f;
#pragma unroll
            for (int j = 0; j < 8; ++j) {
                f.u[j]     = pp[hi * 8 + j];
                f.u[8 + j] = pp[16 + hi * 8 + j];
            }
            pf[mi] = f.h;
        }
#pragma unroll
        for (int ni = 0; ni < 4; ++ni) {
            U16f fv;
#pragma unroll
            for (int j = 0; j < 16; ++j)
                fv.u[j] = base[(size_t)(kc + (j < 8 ? hi * 8 + j : 16 + hi * 8 + (j - 8))) * rs + voffh + ni * 16 + l15];
            acc[0][ni] = __builtin_amdgcn_wmma_f32_16x16x32_f16(
                false, pf[0], false, fv.h, (short)0, acc[0][ni], false, false);
            acc[1][ni] = __builtin_amdgcn_wmma_f32_16x16x32_f16(
                false, pf[1], false, fv.h, (short)0, acc[1][ni], false, false);
        }
        __syncthreads();
    }

#pragma unroll
    for (int mi = 0; mi < 2; ++mi)
#pragma unroll
        for (int ni = 0; ni < 4; ++ni)
#pragma unroll
            for (int r = 0; r < 8; ++r) {
                int srow = qb + mi * 16 + r + hi * 8;
                float o = acc[mi][ni][r] / lr[mi][r];
                attn[((size_t)bIdx * Ss + srow) * Ee + hIdx * 64 + ni * 16 + l15] = f2h(o);
            }
}

// =====================================================================
extern "C" void kernel_launch(void* const* d_in, const int* in_sizes, int n_in,
                              void* d_out, int out_size, void* d_ws, size_t ws_size,
                              hipStream_t stream) {
    (void)in_sizes; (void)n_in; (void)out_size; (void)ws_size;
    const int*   x        = (const int*)d_in[0];
    const float* word_emb = (const float*)d_in[1];
    const float* pos_emb  = (const float*)d_in[2];
    const float* KQV      = (const float*)d_in[3];
    const float* WO       = (const float*)d_in[4];
    const float* W_up     = (const float*)d_in[5];
    const float* b_up     = (const float*)d_in[6];
    const float* W_down   = (const float*)d_in[7];
    const float* b_down   = (const float*)d_in[8];
    const float* ln1_g    = (const float*)d_in[9];
    const float* ln1_b    = (const float*)d_in[10];
    const float* ln2_g    = (const float*)d_in[11];
    const float* ln2_b    = (const float*)d_in[12];
    const float* unemb_b  = (const float*)d_in[13];
    float* logits = (float*)d_out;

    const int M = Bb * Ss;       // 4096 token rows
    size_t off = 0;
    auto carve = [&](size_t bytes) {
        void* p = (char*)d_ws + off;
        off += (bytes + 255) & ~(size_t)255;
        return p;
    };
    u16*   weh   = (u16*)  carve((size_t)Vv * Ee * 2);          // word_embed f16 (V x E) == Bt
    u16*   kqvt  = (u16*)  carve((size_t)Ll * 3 * Ee * Ee * 2); // (3E x E) per layer
    u16*   wot   = (u16*)  carve((size_t)Ll * Ee * Ee * 2);
    u16*   wupt  = (u16*)  carve((size_t)Ll * Ff * Ee * 2);     // (F x E)
    u16*   wdnt  = (u16*)  carve((size_t)Ll * Ee * Ff * 2);     // (E x F)
    float* hf    = (float*)carve((size_t)M * Ee * 4);
    u16*   hh    = (u16*)  carve((size_t)M * Ee * 2);
    u16*   qkvh  = (u16*)  carve((size_t)M * 3 * Ee * 2);
    u16*   attnh = (u16*)  carve((size_t)M * Ee * 2);
    float* zf    = (float*)carve((size_t)M * Ee * 4);
    u16*   ffh   = (u16*)  carve((size_t)M * Ff * 2);

    // weight conversion (word_embed is already N x K for the unembed GEMM)
    {
        int n = Vv * Ee;
        k_conv<<<(n + 255) / 256, 256, 0, stream>>>(word_emb, weh, n);
    }
    for (int l = 0; l < Ll; ++l) {
        int n;
        n = Ee * 3 * Ee;
        k_tconv<<<(n + 255) / 256, 256, 0, stream>>>(KQV + (size_t)l * n, kqvt + (size_t)l * n, Ee, 3 * Ee);
        n = Ee * Ee;
        k_tconv<<<(n + 255) / 256, 256, 0, stream>>>(WO + (size_t)l * n, wot + (size_t)l * n, Ee, Ee);
        n = Ee * Ff;
        k_tconv<<<(n + 255) / 256, 256, 0, stream>>>(W_up + (size_t)l * n, wupt + (size_t)l * n, Ee, Ff);
        n = Ff * Ee;
        k_tconv<<<(n + 255) / 256, 256, 0, stream>>>(W_down + (size_t)l * n, wdnt + (size_t)l * n, Ff, Ee);
    }

    k_embed<<<M, 256, 0, stream>>>(x, word_emb, pos_emb, hf, hh);

    for (int l = 0; l < Ll; ++l) {
        // qkv = h @ KQV  (f16 out)
        k_gemm<<<dim3(3 * Ee / BN, M / BM), 128, 0, stream>>>(
            hh, kqvt + (size_t)l * 3 * Ee * Ee, nullptr, qkvh, nullptr, M, 3 * Ee, Ee, 0);
        // attention
        k_attn<<<dim3(Ss / 32, Bb * Hh), 32, 0, stream>>>(qkvh, attnh);
        // z = attn @ WO (f32 out)
        k_gemm<<<dim3(Ee / BN, M / BM), 128, 0, stream>>>(
            attnh, wot + (size_t)l * Ee * Ee, zf, nullptr, nullptr, M, Ee, Ee, 0);
        // h = LN(z) + h
        k_ln_res<<<M, 256, 0, stream>>>(zf, ln1_g + (size_t)l * Ee, ln1_b + (size_t)l * Ee, hf, hh);
        // ff = relu(h @ W_up + b_up) (f16 out)
        k_gemm<<<dim3(Ff / BN, M / BM), 128, 0, stream>>>(
            hh, wupt + (size_t)l * Ff * Ee, nullptr, ffh, b_up + (size_t)l * Ff, M, Ff, Ee, 1);
        // z = ff @ W_down + b_down (f32 out)
        k_gemm<<<dim3(Ee / BN, M / BM), 128, 0, stream>>>(
            ffh, wdnt + (size_t)l * Ee * Ff, zf, nullptr, b_down + (size_t)l * Ee, M, Ee, Ff, 0);
        // h = LN(z) + h
        k_ln_res<<<M, 256, 0, stream>>>(zf, ln2_g + (size_t)l * Ee, ln2_b + (size_t)l * Ee, hf, hh);
    }

    // logits = h @ word_embed^T + unembed_b  (word_embed is Bt already)
    k_gemm<<<dim3(Vv / BN, M / BM), 128, 0, stream>>>(
        hh, weh, logits, nullptr, unemb_b, M, Vv, Ee, 0);
}
